// MultiheadLatentAttention_20100446945552
// MI455X (gfx1250) — matrix operand order repaired and checked by reading, not compiled, not run on hardware
//
#include <hip/hip_runtime.h>

// ---- problem constants (reference: B,S,DIM=2,2048,2048; H,DK,DR,DC,DCP,DV=16,128,64,512,1024,128)
#define BB   2
#define SS   2048
#define DIM_ 2048
#define HH   16
#define DK_  128
#define DR_  64
#define DC_  512
#define DCP_ 1024
#define DV_  128
#define DKR  192          // DK + DR
#define MTOT (BB*SS)      // 4096 rows

typedef __bf16 v8bf  __attribute__((ext_vector_type(8)));
typedef __bf16 v16bf __attribute__((ext_vector_type(16)));
typedef float  v8f   __attribute__((ext_vector_type(8)));

__device__ __forceinline__ __bf16 f2bf(float f) {
  unsigned u = __builtin_bit_cast(unsigned, f);
  u += 0x7FFFu + ((u >> 16) & 1u);           // round-to-nearest-even
  unsigned short h = (unsigned short)(u >> 16);
  return __builtin_bit_cast(__bf16, h);
}

__device__ __forceinline__ v16bf cat8(v8bf a, v8bf b) {
  return __builtin_shufflevector(a, b, 0,1,2,3,4,5,6,7,8,9,10,11,12,13,14,15);
}

// LDS byte offset (relative to wave LDS base) of a generic pointer to __shared__.
__device__ __forceinline__ unsigned lds_off32(const void* p) {
  return (unsigned)(unsigned long long)(__attribute__((address_space(3))) const char*)p;
}

// CDNA5 async global->LDS copy, 16 bytes per lane; tracked by ASYNCcnt.
__device__ __forceinline__ void async_copy_b128(unsigned lds, const void* g) {
  asm volatile("global_load_async_to_lds_b128 %0, %1, off"
               :: "v"(lds), "v"(g) : "memory");
}
__device__ __forceinline__ void wait_async0() {
  asm volatile("s_wait_asynccnt 0" ::: "memory");
}

// ---------------- fp32 -> bf16 elementwise ----------------
__global__ void __launch_bounds__(256)
k_cvt_bf16(const float* __restrict__ in, __bf16* __restrict__ out, long long n) {
  long long i = (long long)blockIdx.x * blockDim.x + threadIdx.x;
  long long st = (long long)gridDim.x * blockDim.x;
  for (; i < n; i += st) out[i] = f2bf(in[i]);
}

// ---------------- fp32 [K,N] -> bf16 [N,K] (per head via blockIdx.z) ----------------
__global__ void __launch_bounds__(256)
k_transpose_bf16(const float* __restrict__ in, __bf16* __restrict__ out, int K, int N) {
  __shared__ float t[32][33];
  const float* inp = in + (size_t)blockIdx.z * K * N;
  __bf16* outp = out + (size_t)blockIdx.z * K * N;
  int n0 = blockIdx.x * 32, k0 = blockIdx.y * 32;
  int tx = threadIdx.x, ty = threadIdx.y;
#pragma unroll
  for (int i = 0; i < 4; ++i) t[ty + 8*i][tx] = inp[(size_t)(k0 + ty + 8*i) * N + n0 + tx];
  __syncthreads();
#pragma unroll
  for (int i = 0; i < 4; ++i) outp[(size_t)(n0 + ty + 8*i) * K + k0 + tx] = f2bf(t[tx][ty + 8*i]);
}

// ---------------- generic bf16 GEMM: A[M,K] row-major  x  Bt[N,K] row-major ----------------
// out[m,n] -> Out[(m/S)*ob + z*oh + (m%S)*os + (n0..+ocol)*on], bf16 output.
// BM=128 BN=64 BK=32, 256 threads = 8 waves (4x2), each wave 32x32 = 2x2 WMMA tiles.
// Tiles staged via GLOBAL_LOAD_ASYNC_TO_LDS_B128 (ASYNCcnt), double buffered.
__global__ void __launch_bounds__(256)
k_gemm_bf16(const __bf16* __restrict__ A, const __bf16* __restrict__ Bt,
            __bf16* __restrict__ Out, int K, long long wHeadStride,
            long long ob, long long oh, long long os, long long on, long long ocol) {
  constexpr int LDA = 40, LDB = 48;
  __shared__ __align__(64) __bf16 sA[2][128 * LDA];   // 20 KB
  __shared__ __align__(64) __bf16 sB[2][64 * LDB];    // 12 KB
  int tid  = threadIdx.x;
  int lane = tid & 31, wave = tid >> 5;
  int wm = wave >> 1, wn = wave & 1;
  int lm = lane & 15;
  int lo8  = (lane < 16) ? 0 : 8;
  int lo16 = (lane < 16) ? 0 : 16;
  long long m0 = (long long)blockIdx.x * 128;
  long long n0 = (long long)blockIdx.y * 64;
  const __bf16* Bh = Bt + (long long)blockIdx.z * wHeadStride;

  int arow = tid >> 1, ahalf = tid & 1;   // A copy: 16 elems/thread (2 x b128 async)
  int brow = tid >> 2, bq = tid & 3;      // B copy:  8 elems/thread (1 x b128 async)
  const __bf16* Aptr = A  + (m0 + arow) * (long long)K + ahalf * 16;
  const __bf16* Bptr = Bh + (n0 + brow) * (long long)K + bq * 8;

  unsigned laBase[2], lbBase[2];
#pragma unroll
  for (int s2 = 0; s2 < 2; ++s2) {
    laBase[s2] = lds_off32(&sA[s2][arow * LDA + ahalf * 16]);
    lbBase[s2] = lds_off32(&sB[s2][brow * LDB + bq * 8]);
  }

  v8f acc[2][2] = {};
  int nk = K >> 5;

  // prologue: async-stage k-step 0
  async_copy_b128(laBase[0],      Aptr);
  async_copy_b128(laBase[0] + 16, Aptr + 8);
  async_copy_b128(lbBase[0],      Bptr);
  wait_async0();
  __syncthreads();

  for (int kt = 0; kt < nk; ++kt) {
    int cur = kt & 1, nxt = cur ^ 1;
    if (kt + 1 < nk) {
      const __bf16* ap = Aptr + (long long)(kt + 1) * 32;
      const __bf16* bp = Bptr + (long long)(kt + 1) * 32;
      async_copy_b128(laBase[nxt],      ap);
      async_copy_b128(laBase[nxt] + 16, ap + 8);
      async_copy_b128(lbBase[nxt],      bp);
      if (kt + 2 < nk) {
        __builtin_prefetch(Aptr + (long long)(kt + 2) * 32, 0, 1);  // global_prefetch_b8
        __builtin_prefetch(Bptr + (long long)(kt + 2) * 32, 0, 1);
      }
    }
    v16bf af[2], bf[2];
#pragma unroll
    for (int sm = 0; sm < 2; ++sm) {
      int row = wm * 32 + sm * 16 + lm;
      af[sm] = cat8(*(const v8bf*)&sA[cur][row * LDA + lo16],
                    *(const v8bf*)&sA[cur][row * LDA + lo16 + 8]);
    }
#pragma unroll
    for (int sn = 0; sn < 2; ++sn) {
      int col = wn * 32 + sn * 16 + lm;
      bf[sn] = *(const v16bf*)&sB[cur][col * LDB + lo16];
    }
#pragma unroll
    for (int sm = 0; sm < 2; ++sm)
#pragma unroll
      for (int sn = 0; sn < 2; ++sn)
        acc[sm][sn] = __builtin_amdgcn_wmma_f32_16x16x32_bf16(
            false, af[sm], false, bf[sn], (short)0, acc[sm][sn], false, false);
    // my async writes to nxt must be LDS-visible to all waves before next iter
    wait_async0();
    __syncthreads();
  }

  // epilogue: C layout -> strided bf16 store
#pragma unroll
  for (int sm = 0; sm < 2; ++sm)
#pragma unroll
    for (int sn = 0; sn < 2; ++sn)
#pragma unroll
      for (int r = 0; r < 8; ++r) {
        long long mrow = m0 + wm * 32 + sm * 16 + r + ((lane < 16) ? 0 : 8);
        long long ncol = n0 + wn * 32 + sn * 16 + lm + ocol;
        long long bidx = mrow / SS, srow = mrow % SS;
        Out[bidx * ob + (long long)blockIdx.z * oh + srow * os + ncol * on] =
            f2bf(acc[sm][sn][r]);
      }
}

// ---------------- flash attention: Q[B,H,S,192], K[B,H,S,192], Vt[B,H,128,S] -> out[B,S,H*128]
__global__ void __launch_bounds__(128)
k_mla_attn(const __bf16* __restrict__ Q, const __bf16* __restrict__ Kb,
           const __bf16* __restrict__ Vt, float* __restrict__ Out) {
  int tid = threadIdx.x;
  int lane = tid & 31, wave = tid >> 5;
  int lm = lane & 15;
  int lo8  = (lane < 16) ? 0 : 8;
  int lo16 = (lane < 16) ? 0 : 16;
  int rbase = (lane < 16) ? 0 : 8;
  int bh = blockIdx.y;
  int q0 = blockIdx.x * 64 + wave * 16;
  const __bf16* qp = Q  + (long long)bh * SS * DKR;
  const __bf16* kp = Kb + (long long)bh * SS * DKR;
  const __bf16* vp = Vt + (long long)bh * DV_ * SS;

  // Q A-fragments, 6 chunks of K=32 over d=192
  v16bf qf[6];
#pragma unroll
  for (int c = 0; c < 6; ++c) {
    const __bf16* p = qp + (long long)(q0 + lm) * DKR + c * 32;
    qf[c] = cat8(*(const v8bf*)(p + lo16), *(const v8bf*)(p + lo16 + 8));
  }

  v8f o[8] = {};
  float mrow[8], lrow[8];
#pragma unroll
  for (int r = 0; r < 8; ++r) { mrow[r] = -1e30f; lrow[r] = 0.f; }

  __shared__ __align__(64) __bf16 pbuf[4][16 * 32];   // per-wave P scratch
  __bf16* P = pbuf[wave];

  const float scale = 0.07216878364870323f;  // 1/sqrt(192)
  int nblk = (q0 + 15) / 32 + 1;             // causal key-block count

  for (int blk = 0; blk < nblk; ++blk) {
    int t0 = blk * 32;
    v8f s[2];
#pragma unroll
    for (int kt2 = 0; kt2 < 2; ++kt2) {
      v8f acc = {};
      long long trow = t0 + kt2 * 16 + lm;
      const __bf16* kpp = kp + trow * DKR;
#pragma unroll
      for (int c = 0; c < 6; ++c) {
        v16bf kf = *(const v16bf*)(kpp + c * 32 + lo16);
        acc = __builtin_amdgcn_wmma_f32_16x16x32_bf16(
            false, qf[c], false, kf, (short)0, acc, false, false);
      }
      s[kt2] = acc * scale;
    }
    if (t0 + 31 > q0) {  // diagonal block: causal mask
#pragma unroll
      for (int kt2 = 0; kt2 < 2; ++kt2) {
        int col = t0 + kt2 * 16 + lm;
#pragma unroll
        for (int r = 0; r < 8; ++r)
          if (col > q0 + r + rbase) s[kt2][r] = -1e30f;
      }
    }
    // online softmax, per C-layout row r (row's 16 cols live across a 16-lane group)
#pragma unroll
    for (int r = 0; r < 8; ++r) {
      float v = fmaxf(s[0][r], s[1][r]);
      v = fmaxf(v, __shfl_xor(v, 1));
      v = fmaxf(v, __shfl_xor(v, 2));
      v = fmaxf(v, __shfl_xor(v, 4));
      v = fmaxf(v, __shfl_xor(v, 8));
      float mn = fmaxf(mrow[r], v);
      float alpha = __expf(mrow[r] - mn);
      float p0 = __expf(s[0][r] - mn);
      float p1 = __expf(s[1][r] - mn);
      float rs = p0 + p1;
      rs += __shfl_xor(rs, 1);
      rs += __shfl_xor(rs, 2);
      rs += __shfl_xor(rs, 4);
      rs += __shfl_xor(rs, 8);
      lrow[r] = lrow[r] * alpha + rs;
      mrow[r] = mn;
#pragma unroll
      for (int vt = 0; vt < 8; ++vt) o[vt][r] *= alpha;
      int prow = r + rbase;
      P[prow * 32 + lm]      = f2bf(p0);   // C-layout -> LDS (wave-private, DS in-order)
      P[prow * 32 + 16 + lm] = f2bf(p1);
    }
    // reload P in A-fragment layout, then P(16x32) x V(32x128)
    v16bf pf = cat8(*(const v8bf*)&P[lm * 32 + lo16],
                    *(const v8bf*)&P[lm * 32 + lo16 + 8]);
#pragma unroll
    for (int vt = 0; vt < 8; ++vt) {
      v16bf vf = *(const v16bf*)(vp + (long long)(vt * 16 + lm) * SS + t0 + lo16);
      o[vt] = __builtin_amdgcn_wmma_f32_16x16x32_bf16(
          false, pf, false, vf, (short)0, o[vt], false, false);
    }
  }

  int b = bh >> 4, h = bh & 15;  // H = 16
#pragma unroll
  for (int r = 0; r < 8; ++r) {
    float inv = 1.0f / lrow[r];
    long long row = q0 + r + rbase;
    float* op = Out + ((long long)b * SS + row) * (HH * DV_) + (long long)h * DV_;
#pragma unroll
    for (int vt = 0; vt < 8; ++vt) op[vt * 16 + lm] = o[vt][r] * inv;
  }
}

// ---------------- launcher ----------------
extern "C" void kernel_launch(void* const* d_in, const int* in_sizes, int n_in,
                              void* d_out, int out_size, void* d_ws, size_t ws_size,
                              hipStream_t stream) {
  (void)in_sizes; (void)n_in; (void)out_size; (void)ws_size;
  const float* x   = (const float*)d_in[0];
  const float* Wc  = (const float*)d_in[1];
  const float* Wcp = (const float*)d_in[2];
  const float* Wqc = (const float*)d_in[3];
  const float* Wqr = (const float*)d_in[4];
  const float* Wkc = (const float*)d_in[5];
  const float* Wkr = (const float*)d_in[6];
  const float* Wv  = (const float*)d_in[7];
  float* out = (float*)d_out;

  char* base = (char*)d_ws;
  size_t off = 0;
  auto carve = [&](size_t elems) -> __bf16* {
    __bf16* p = (__bf16*)(base + off);
    off += ((elems * 2) + 255) & ~(size_t)255;
    return p;
  };
  __bf16* xb   = carve((size_t)MTOT * DIM_);
  __bf16* WcT  = carve((size_t)DC_  * DIM_);
  __bf16* WcpT = carve((size_t)DCP_ * DIM_);
  __bf16* WqcT = carve((size_t)HH * DK_ * DCP_);
  __bf16* WqrT = carve((size_t)HH * DR_ * DCP_);
  __bf16* WkcT = carve((size_t)HH * DK_ * DC_);
  __bf16* WkrT = carve((size_t)DR_ * DIM_);
  __bf16* WvT  = carve((size_t)HH * DV_ * DC_);
  __bf16* cb   = carve((size_t)MTOT * DC_);
  __bf16* cpb  = carve((size_t)MTOT * DCP_);
  __bf16* qb   = carve((size_t)BB * HH * SS * DKR);
  __bf16* kb   = carve((size_t)BB * HH * SS * DKR);
  __bf16* vtb  = carve((size_t)BB * HH * DV_ * SS);

  // 1) bf16 conversions (+ transposed weights)
  k_cvt_bf16<<<4096, 256, 0, stream>>>(x, xb, (long long)MTOT * DIM_);
  dim3 tb(32, 8);
  k_transpose_bf16<<<dim3(DC_/32,  DIM_/32, 1),  tb, 0, stream>>>(Wc,  WcT,  DIM_, DC_);
  k_transpose_bf16<<<dim3(DCP_/32, DIM_/32, 1),  tb, 0, stream>>>(Wcp, WcpT, DIM_, DCP_);
  k_transpose_bf16<<<dim3(DK_/32,  DCP_/32, HH), tb, 0, stream>>>(Wqc, WqcT, DCP_, DK_);
  k_transpose_bf16<<<dim3(DR_/32,  DCP_/32, HH), tb, 0, stream>>>(Wqr, WqrT, DCP_, DR_);
  k_transpose_bf16<<<dim3(DK_/32,  DC_/32,  HH), tb, 0, stream>>>(Wkc, WkcT, DC_,  DK_);
  k_transpose_bf16<<<dim3(DR_/32,  DIM_/32, 1),  tb, 0, stream>>>(Wkr, WkrT, DIM_, DR_);
  k_transpose_bf16<<<dim3(DV_/32,  DC_/32,  HH), tb, 0, stream>>>(Wv,  WvT,  DC_,  DV_);

  long long qob = (long long)HH * SS * DKR, qoh = (long long)SS * DKR;

  // 2) projection GEMMs
  // c = x @ Wc -> [B,S,DC]
  k_gemm_bf16<<<dim3(MTOT/128, DC_/64, 1), 256, 0, stream>>>(
      xb, WcT, cb, DIM_, 0, (long long)SS * DC_, 0, DC_, 1, 0);
  // cp = x @ Wcp -> [B,S,DCP]
  k_gemm_bf16<<<dim3(MTOT/128, DCP_/64, 1), 256, 0, stream>>>(
      xb, WcpT, cpb, DIM_, 0, (long long)SS * DCP_, 0, DCP_, 1, 0);
  // q_c -> q[...,0:128]
  k_gemm_bf16<<<dim3(MTOT/128, DK_/64, HH), 256, 0, stream>>>(
      cpb, WqcT, qb, DCP_, (long long)DK_ * DCP_, qob, qoh, DKR, 1, 0);
  // q_r -> q[...,128:192]
  k_gemm_bf16<<<dim3(MTOT/128, DR_/64, HH), 256, 0, stream>>>(
      cpb, WqrT, qb, DCP_, (long long)DR_ * DCP_, qob, qoh, DKR, 1, DK_);
  // k_c -> k[...,0:128]
  k_gemm_bf16<<<dim3(MTOT/128, DK_/64, HH), 256, 0, stream>>>(
      cb, WkcT, kb, DC_, (long long)DK_ * DC_, qob, qoh, DKR, 1, 0);
  // k_r broadcast to all heads -> k[...,128:192] (head stride 0 on weights)
  k_gemm_bf16<<<dim3(MTOT/128, DR_/64, HH), 256, 0, stream>>>(
      xb, WkrT, kb, DIM_, 0, qob, qoh, DKR, 1, DK_);
  // v stored transposed -> Vt[B,H,DV,S]  (os=1, on=S)
  k_gemm_bf16<<<dim3(MTOT/128, DV_/64, HH), 256, 0, stream>>>(
      cb, WvT, vtb, DC_, (long long)DV_ * DC_,
      (long long)HH * DV_ * SS, (long long)DV_ * SS, 1, SS, 0);

  // 3) causal flash attention -> out [B,S,H*DV] fp32
  k_mla_attn<<<dim3(SS/64, BB*HH), 128, 0, stream>>>(qb, kb, vtb, out);
}
